// TopKAuto_49804440764452
// MI455X (gfx1250) — matrix operand order repaired and checked by reading, not compiled, not run on hardware
//
#include <hip/hip_runtime.h>

#define BB 4096
#define DD 1024
#define HH 16384
#define KK 64
#define LN_EPS 1e-5f

typedef __bf16 bf16_t;
typedef __attribute__((ext_vector_type(16))) __bf16 v16bf;
typedef __attribute__((ext_vector_type(8)))  __bf16 v8bf;
typedef __attribute__((ext_vector_type(4)))  __bf16 v4bf;
typedef __attribute__((ext_vector_type(8)))  float  v8f;
typedef __attribute__((ext_vector_type(4)))  unsigned int u32x4;
typedef __attribute__((ext_vector_type(8)))  int i32x8;
typedef __attribute__((ext_vector_type(4)))  int i32x4;

// LDS tile geometry: 128 rows x 32 bf16 (64B) + 16B TDM pad per row -> 80B pitch.
#define TILE_ROW_E 40            // row pitch in bf16 elements (80 bytes)
#define TILE_ELEMS (128 * TILE_ROW_E)

// ---------------------------------------------------------------------------
// Kernel 1: LayerNorm over D=1024 per row, emit bf16 row-major [B, D]
// ---------------------------------------------------------------------------
__global__ void ln_bf16_kernel(const float* __restrict__ x,
                               unsigned short* __restrict__ lnb_)
{
    bf16_t* lnb = reinterpret_cast<bf16_t*>(lnb_);
    const int row = blockIdx.x;
    const int tid = threadIdx.x;                       // 256 threads, 4 floats each
    const float4 v = reinterpret_cast<const float4*>(x + (size_t)row * DD)[tid];

    float s  = v.x + v.y + v.z + v.w;
    float s2 = v.x * v.x + v.y * v.y + v.z * v.z + v.w * v.w;

    __shared__ float rs[256];
    __shared__ float rq[256];
    rs[tid] = s; rq[tid] = s2;
    __syncthreads();
    #pragma unroll
    for (int off = 128; off > 0; off >>= 1) {
        if (tid < off) { rs[tid] += rs[tid + off]; rq[tid] += rq[tid + off]; }
        __syncthreads();
    }
    const float mean = rs[0] * (1.0f / (float)DD);
    const float var  = rq[0] * (1.0f / (float)DD) - mean * mean;
    const float inv  = rsqrtf(var + LN_EPS);

    v4bf o;
    o[0] = (bf16_t)((v.x - mean) * inv);
    o[1] = (bf16_t)((v.y - mean) * inv);
    o[2] = (bf16_t)((v.z - mean) * inv);
    o[3] = (bf16_t)((v.w - mean) * inv);
    reinterpret_cast<v4bf*>(lnb + (size_t)row * DD)[tid] = o;
}

// ---------------------------------------------------------------------------
// Kernel 2: W_enc [D, H] f32  ->  W_t [H, D] bf16 (transpose + convert)
// ---------------------------------------------------------------------------
__global__ void wenc_transpose_kernel(const float* __restrict__ W,
                                      unsigned short* __restrict__ wt_)
{
    bf16_t* wt = reinterpret_cast<bf16_t*>(wt_);
    __shared__ bf16_t tile[32][33];
    const int tx = threadIdx.x;                        // 0..31
    const int ty = threadIdx.y;                        // 0..7
    const int h0 = blockIdx.x * 32;
    const int d0 = blockIdx.y * 32;

    #pragma unroll
    for (int i = 0; i < 4; ++i) {
        const int dl = ty + i * 8;
        tile[dl][tx] = (bf16_t)W[(size_t)(d0 + dl) * HH + (h0 + tx)];
    }
    __syncthreads();
    #pragma unroll
    for (int i = 0; i < 4; ++i) {
        const int hl = ty + i * 8;
        wt[(size_t)(h0 + hl) * DD + (d0 + tx)] = tile[tx][hl];
    }
}

// ---------------------------------------------------------------------------
// TDM helper: issue a 2D tensor_load_to_lds for a 16-row x 32-element bf16
// tile (stride 1024 elements) with LDS padding 16B per 64B row (80B pitch).
// D# fields per CDNA5 ISA. This toolchain's
// builtin takes 6 args: (g0 u32x4, g1 i32x8, g2 i32x4, g3 i32x4, i32x8, cpol).
// ---------------------------------------------------------------------------
__device__ __forceinline__ void tdm_load_slice(unsigned int lds_off,
                                               const bf16_t* gptr,
                                               i32x8 g1)
{
    const unsigned long long ga = (unsigned long long)(size_t)gptr;
    u32x4 g0;
    g0[0] = 1u;                                              // count=1, user mode
    g0[1] = lds_off;                                         // lds_addr (bytes)
    g0[2] = (unsigned int)ga;                                // global_addr[31:0]
    g0[3] = (unsigned int)((ga >> 32) & 0x01FFFFFFu)         // global_addr[56:32]
          | (2u << 30);                                      // type = 2 ("image")
    i32x4 z4 = {0, 0, 0, 0};                                 // 2D: groups 2/3 unused
    i32x8 z8 = {0, 0, 0, 0, 0, 0, 0, 0};
    __builtin_amdgcn_tensor_load_to_lds(g0, g1, z4, z4, z8, 0);
}

// ---------------------------------------------------------------------------
// Kernel 3: encoder GEMM via v_wmma_f32_16x16x32_bf16 with TDM-staged,
// double-buffered LDS tiles. Block = 8 wave32s (4 M-waves x 2 N-waves),
// wave tile 32x64, block tile 128x128, K-step 32.
// ---------------------------------------------------------------------------
__global__ void enc_wmma_kernel(const unsigned short* __restrict__ lnb_,
                                const unsigned short* __restrict__ wt_,
                                const float* __restrict__ b_enc,
                                float* __restrict__ feats)
{
    const bf16_t* lnb = reinterpret_cast<const bf16_t*>(lnb_);
    const bf16_t* wt  = reinterpret_cast<const bf16_t*>(wt_);

    __shared__ bf16_t lA[2][TILE_ELEMS];    // 2 x 10KB
    __shared__ bf16_t lB[2][TILE_ELEMS];    // 2 x 10KB

    const int lane = threadIdx.x & 31;
    const int wv   = __builtin_amdgcn_readfirstlane(threadIdx.x >> 5); // scalar wave id
    const int mw   = wv & 3;                         // 0..3
    const int nw   = wv >> 2;                        // 0..1
    const int l16  = lane & 15;
    const int half = lane >> 4;

    const int rowBase0 = blockIdx.y * 128;           // block M origin
    const int colBase0 = blockIdx.x * 128;           // block N origin
    const int rowBase  = rowBase0 + mw * 32;
    const int colBase  = colBase0 + nw * 64;

    // D# group 1 (constant): workgroup_mask=0, data_size=1 (2B),
    // pad_enable=1, pad_interval=3 (16 dwords), pad_amount=3 (4 dwords),
    // tensor_dim0=32, tensor_dim1=16, tile_dim0=32, tile_dim1=16,
    // tensor_dim0_stride=1024.
    i32x8 g1;
    g1[0] = (1 << 16) | (1 << 20) | (3 << 22) | (3 << 25);
    g1[1] = (32 & 0xFFFF) << 16;                     // tensor_dim0[15:0]
    g1[2] = (16 & 0xFFFF) << 16;                     // tensor_dim0[31:16]=0 | tensor_dim1[15:0]
    g1[3] = (32 & 0xFFFF) << 16;                     // tensor_dim1[31:16]=0 | tile_dim0
    g1[4] = 16;                                      // tile_dim1 | tile_dim2=0
    g1[5] = 1024;                                    // tensor_dim0_stride[31:0]
    g1[6] = 0;                                       // stride hi | dim1_stride lo
    g1[7] = 0;

    // Per-wave DMA slice: 16 rows of each 128-row tile.
    const unsigned int ldsA[2] = {
        (unsigned int)(size_t)(&lA[0][0]) + (unsigned int)(wv * 16 * 80),
        (unsigned int)(size_t)(&lA[1][0]) + (unsigned int)(wv * 16 * 80) };
    const unsigned int ldsB[2] = {
        (unsigned int)(size_t)(&lB[0][0]) + (unsigned int)(wv * 16 * 80),
        (unsigned int)(size_t)(&lB[1][0]) + (unsigned int)(wv * 16 * 80) };
    const bf16_t* gA = lnb + (size_t)(rowBase0 + wv * 16) * DD;   // + kt
    const bf16_t* gB = wt  + (size_t)(colBase0 + wv * 16) * DD;   // + kt

    v8f acc[2][4];
    #pragma unroll
    for (int m = 0; m < 2; ++m)
        #pragma unroll
        for (int n = 0; n < 4; ++n)
            acc[m][n] = (v8f){0.f, 0.f, 0.f, 0.f, 0.f, 0.f, 0.f, 0.f};

    // Prime buffer 0 with the kt=0 tiles.
    tdm_load_slice(ldsA[0], gA, g1);
    tdm_load_slice(ldsB[0], gB, g1);
    __builtin_amdgcn_s_wait_tensorcnt(0);
    __syncthreads();

    int buf = 0;
    for (int kt = 0; kt < DD; kt += 32) {
        const bool more = (kt + 32) < DD;
        if (more) {                                   // uniform branch
            tdm_load_slice(ldsA[buf ^ 1], gA + kt + 32, g1);
            tdm_load_slice(ldsB[buf ^ 1], gB + kt + 32, g1);
        }

        const bf16_t* cA = lA[buf];
        const bf16_t* cB = lB[buf];

        // A fragment: 16-bit A layout -> half=0: K0-7 & K16-23, half=1: K8-15 & K24-31
        v16bf afrag[2];
        #pragma unroll
        for (int m = 0; m < 2; ++m) {
            const int mloc = mw * 32 + m * 16 + l16;
            const bf16_t* ar = cA + mloc * TILE_ROW_E;
            v8bf lo = *reinterpret_cast<const v8bf*>(ar + half * 8);
            v8bf hi = *reinterpret_cast<const v8bf*>(ar + half * 8 + 16);
            afrag[m] = __builtin_shufflevector(lo, hi,
                0, 1, 2, 3, 4, 5, 6, 7, 8, 9, 10, 11, 12, 13, 14, 15);
        }
        #pragma unroll
        for (int n = 0; n < 4; ++n) {
            const int nloc = nw * 64 + n * 16 + l16;
            const bf16_t* br = cB + nloc * TILE_ROW_E;
            v8bf lo = *reinterpret_cast<const v8bf*>(br + half * 8);
            v8bf hi = *reinterpret_cast<const v8bf*>(br + half * 8 + 16);
            v16bf bfrag = __builtin_shufflevector(lo, hi,
                0, 1, 2, 3, 4, 5, 6, 7, 8, 9, 10, 11, 12, 13, 14, 15);

            acc[0][n] = __builtin_amdgcn_wmma_f32_16x16x32_bf16(
                false, afrag[0], false, bfrag, (short)0, acc[0][n], false, false);
            acc[1][n] = __builtin_amdgcn_wmma_f32_16x16x32_bf16(
                false, afrag[1], false, bfrag, (short)0, acc[1][n], false, false);
        }

        if (more) {                                   // uniform branch
            __builtin_amdgcn_s_wait_tensorcnt(0);     // next tiles landed (this wave)
            __syncthreads();                          // ...and every other wave's
        }
        buf ^= 1;
    }

    // C layout: VGPR r, lanes 0-15 -> M=r, lanes 16-31 -> M=8+r; N = lane&15.
    #pragma unroll
    for (int n = 0; n < 4; ++n) {
        const int col = colBase + n * 16 + l16;
        const float be = b_enc[col];
        #pragma unroll
        for (int m = 0; m < 2; ++m) {
            const int r0 = rowBase + m * 16 + half * 8;
            #pragma unroll
            for (int r = 0; r < 8; ++r)
                feats[(size_t)(r0 + r) * HH + col] = acc[m][n][r] + be;
        }
    }
}

// ---------------------------------------------------------------------------
// Kernel 4: per-row top-64 + ReLU scatter (in place) + sparse decoder.
// ---------------------------------------------------------------------------
__global__ void topk_decode_kernel(float* __restrict__ sparse,
                                   const float* __restrict__ W_dec,
                                   const float* __restrict__ b_dec,
                                   float* __restrict__ recon)
{
    extern __shared__ float fv[];                      // HH floats (64 KB)
    __shared__ float rmax[256];
    __shared__ int   rarg[256];
    __shared__ float topv[KK];
    __shared__ int   topi[KK];

    const int row = blockIdx.x;
    const int tid = threadIdx.x;
    float* frow = sparse + (size_t)row * HH;

    for (int i = tid; i < HH; i += 256) fv[i] = frow[i];
    __syncthreads();

    for (int k = 0; k < KK; ++k) {
        float best = -3.4028235e38f; int bi = 0;
        for (int i = tid; i < HH; i += 256) {
            const float v = fv[i];
            if (v > best) { best = v; bi = i; }
        }
        rmax[tid] = best; rarg[tid] = bi;
        __syncthreads();
        for (int off = 128; off > 0; off >>= 1) {
            if (tid < off && rmax[tid + off] > rmax[tid]) {
                rmax[tid] = rmax[tid + off];
                rarg[tid] = rarg[tid + off];
            }
            __syncthreads();
        }
        if (tid == 0) {
            topv[k] = rmax[0];
            topi[k] = rarg[0];
            fv[rarg[0]] = -3.4028235e38f;
        }
        __syncthreads();
    }

    for (int i = tid; i < HH; i += 256) frow[i] = 0.0f;
    __syncthreads();
    if (tid < KK) {
        const float v = topv[tid];
        if (v > 0.0f) frow[topi[tid]] = v;
    }

    const int d = tid * 4;
    float4 accv = *reinterpret_cast<const float4*>(b_dec + d);
    for (int k = 0; k < KK; ++k) {
        const float a = fmaxf(topv[k], 0.0f);
        const float4 wv =
            *reinterpret_cast<const float4*>(W_dec + (size_t)topi[k] * DD + d);
        accv.x += a * wv.x; accv.y += a * wv.y;
        accv.z += a * wv.z; accv.w += a * wv.w;
    }
    *reinterpret_cast<float4*>(recon + (size_t)row * DD + d) = accv;
}

// ---------------------------------------------------------------------------
// Host launcher
// ---------------------------------------------------------------------------
extern "C" void kernel_launch(void* const* d_in, const int* in_sizes, int n_in,
                              void* d_out, int out_size, void* d_ws, size_t ws_size,
                              hipStream_t stream)
{
    (void)in_sizes; (void)n_in; (void)out_size; (void)ws_size;

    const float* x     = (const float*)d_in[0];
    const float* W_enc = (const float*)d_in[1];
    const float* W_dec = (const float*)d_in[2];
    const float* b_enc = (const float*)d_in[3];
    const float* b_dec = (const float*)d_in[4];

    float* recon  = (float*)d_out;                     // [B, D]
    float* sparse = recon + (size_t)BB * DD;           // [B, H] (also feats buf)

    unsigned short* lnb = (unsigned short*)d_ws;       // [B, D] bf16 (8 MB)
    unsigned short* wt  = lnb + (size_t)BB * DD;       // [H, D] bf16 (32 MB)

    ln_bf16_kernel<<<BB, 256, 0, stream>>>(x, lnb);
    wenc_transpose_kernel<<<dim3(HH / 32, DD / 32), dim3(32, 8), 0, stream>>>(W_enc, wt);
    enc_wmma_kernel<<<dim3(HH / 128, BB / 128), 256, 0, stream>>>(lnb, wt, b_enc, sparse);
    topk_decode_kernel<<<BB, 256, HH * sizeof(float), stream>>>(sparse, W_dec, b_dec, recon);
}
